// all_atom_view_graph_opt_54030688584317
// MI455X (gfx1250) — compile-verified, matrix operand order checked by reading, not run on hardware
//
#include <hip/hip_runtime.h>

// ---------------------------------------------------------------- constants
#define N_NODES  18432
#define E_TOT    368640
#define HID      128
#define SUM_LIG  2048
#define SUM_PRO  16384
#define C_LIG    512
#define C_PRO    4096
#define N_LAYERS 4

// output flat offsets (x_lig_cg, x_pro_cg, pos_lig, pos_pro)
#define OUT_XPRO  (C_LIG*HID)                    // 65536
#define OUT_PLIG  (OUT_XPRO + C_PRO*HID)         // 589824
#define OUT_PPRO  (OUT_PLIG + SUM_LIG*3)         // 595968

// ---------------------------------------------------------------- types
typedef __attribute__((ext_vector_type(16))) __bf16 v16bf;
typedef __attribute__((ext_vector_type(8)))  float  v8f;

struct alignas(16) U128 { unsigned int w[4]; };

union FragU {
    v16bf          v;
    unsigned short u[16];
    U128           q[2];
};

// ---------------------------------------------------------------- helpers
__device__ __forceinline__ unsigned short f2bf(float f) {
    unsigned u = __float_as_uint(f);
    u += 0x7fffu + ((u >> 16) & 1u);          // round-to-nearest-even
    return (unsigned short)(u >> 16);
}

__device__ __forceinline__ v8f wmma_bf16(v16bf a, v16bf b, v8f c) {
    return __builtin_amdgcn_wmma_f32_16x16x32_bf16(false, a, false, b, (short)0, c, false, false);
}

__device__ __forceinline__ v8f silu8(v8f x) {
    v8f r;
#pragma unroll
    for (int i = 0; i < 8; ++i) { float v = x[i]; r[i] = v / (1.0f + __expf(-v)); }
    return r;
}

// B-fragment load: frag stored as 32 lanes x 16 bf16, lane-contiguous
__device__ __forceinline__ v16bf load_bfrag(const unsigned short* frag, int lane) {
    FragU f;
    const U128* p = (const U128*)(frag + lane * 16);
    f.q[0] = p[0]; f.q[1] = p[1];
    return f.v;
}

// A-fragment from a bf16 row pointer (row-major, 128 wide); k0 = kc*32 + 8*half
__device__ __forceinline__ v16bf load_afrag_bf(const unsigned short* rowp, int k0) {
    FragU f;
    f.q[0] = *(const U128*)(rowp + k0);
    f.q[1] = *(const U128*)(rowp + k0 + 16);
    return f.v;
}

// A-fragment from an f32 row pointer, converting to bf16
__device__ __forceinline__ v16bf load_afrag_f32(const float* rowp, int k0) {
    FragU f;
#pragma unroll
    for (int i = 0; i < 8; ++i) {
        f.u[i]     = f2bf(rowp[k0 + i]);
        f.u[8 + i] = f2bf(rowp[k0 + 16 + i]);
    }
    return f.v;
}

// monotonic float <-> uint for atomicMax-based segment max
__device__ __forceinline__ unsigned fmap(float f) {
    int s = __float_as_int(f);
    unsigned u = (unsigned)s;
    return (s < 0) ? ~u : (u | 0x80000000u);
}
__device__ __forceinline__ float funmap(unsigned u) {
    int s = (u & 0x80000000u) ? (int)(u & 0x7fffffffu) : (int)~u;
    return __int_as_float(s);
}

// ---------------------------------------------------------------- weight prep
// src: f32 KxN (N=128) row-major; dst: frags[kc*8+nt], 512 bf16 each,
// element(lane,s) = src[kc*32 + kmap(s,lane)][nt*16 + (lane&15)]
__global__ void prep_frag_kernel(const float* __restrict__ src, int Ksrc,
                                 unsigned short* __restrict__ dst) {
    int f  = blockIdx.x;
    int kc = f >> 3, nt = f & 7;
    for (int e = threadIdx.x; e < 512; e += blockDim.x) {
        int lane = e >> 4, s = e & 15;
        int k = kc * 32 + ((s < 8) ? s : s + 8) + ((lane < 16) ? 0 : 8);
        int n = nt * 16 + (lane & 15);
        float v = (k < Ksrc) ? src[k * HID + n] : 0.0f;
        dst[f * 512 + e] = f2bf(v);
    }
}

// ---------------------------------------------------------------- degree
__global__ void degree_kernel(const int* __restrict__ eidx, float* __restrict__ deg) {
    int e = blockIdx.x * blockDim.x + threadIdx.x;
    if (e < E_TOT) atomicAdd(&deg[eidx[e]], 1.0f);
}

// ---------------------------------------------------------------- node embed
// h = silu(x@lin_w+lin_b) @ emb_in_w + emb_in_b
__global__ __launch_bounds__(256) void embed_kernel(
    const float* __restrict__ x,
    const unsigned short* __restrict__ fr_lin, const float* __restrict__ lin_b,
    const unsigned short* __restrict__ fr_emb, const float* __restrict__ emb_b,
    float* __restrict__ h, unsigned short* __restrict__ hbf) {
    __shared__ alignas(16) unsigned short lds_m[8][16 * HID];
    const int wave = threadIdx.x >> 5, lane = threadIdx.x & 31;
    const int M = lane & 15, half = lane >> 4, koff = half * 8;
    const int n0 = (blockIdx.x * 8 + wave) * 16;
    const float* rowx = x + (size_t)(n0 + M) * 44;

    v16bf a[2];
#pragma unroll
    for (int kc = 0; kc < 2; ++kc) {
        FragU f;
#pragma unroll
        for (int i = 0; i < 8; ++i) {
            int k = kc * 32 + koff + i;
            f.u[i]     = (k      < 44) ? f2bf(rowx[k])      : (unsigned short)0;
            f.u[8 + i] = (k + 16 < 44) ? f2bf(rowx[k + 16]) : (unsigned short)0;
        }
        a[kc] = f.v;
    }
#pragma unroll
    for (int nt = 0; nt < 8; ++nt) {
        v8f acc; float b = lin_b[nt * 16 + M];
#pragma unroll
        for (int v = 0; v < 8; ++v) acc[v] = b;
#pragma unroll
        for (int kc = 0; kc < 2; ++kc)
            acc = wmma_bf16(a[kc], load_bfrag(fr_lin + (kc * 8 + nt) * 512, lane), acc);
        acc = silu8(acc);
#pragma unroll
        for (int v = 0; v < 8; ++v)
            lds_m[wave][(v + half * 8) * HID + nt * 16 + M] = f2bf(acc[v]);
    }
    __syncthreads();
    v16bf a2[4];
#pragma unroll
    for (int kc = 0; kc < 4; ++kc)
        a2[kc] = load_afrag_bf(&lds_m[wave][M * HID], kc * 32 + koff);
#pragma unroll
    for (int nt = 0; nt < 8; ++nt) {
        v8f acc; float b = emb_b[nt * 16 + M];
#pragma unroll
        for (int v = 0; v < 8; ++v) acc[v] = b;
#pragma unroll
        for (int kc = 0; kc < 4; ++kc)
            acc = wmma_bf16(a2[kc], load_bfrag(fr_emb + (kc * 8 + nt) * 512, lane), acc);
#pragma unroll
        for (int v = 0; v < 8; ++v) {
            size_t idx = (size_t)(n0 + v + half * 8) * HID + nt * 16 + M;
            h[idx] = acc[v];
            hbf[idx] = f2bf(acc[v]);
        }
    }
}

// ---------------------------------------------------------------- edge kernel
__global__ __launch_bounds__(256) void edge_kernel(
    const unsigned short* __restrict__ hbf, const float* __restrict__ pos,
    const int* __restrict__ eidx, const float* __restrict__ edge_attr,
    const float* __restrict__ ew1_full,   // layer base of [261][128] f32 (extras rows 256..260)
    const float* __restrict__ eb1, const float* __restrict__ eb2,
    const float* __restrict__ cb1, const float* __restrict__ cw2,
    const unsigned short* __restrict__ fr_ew1,
    const unsigned short* __restrict__ fr_ew2,
    const unsigned short* __restrict__ fr_cw1,
    float* __restrict__ agg, float* __restrict__ posnum) {
    __shared__ alignas(16) unsigned short lds_m[8][16 * HID];
    __shared__ int   s_row[8][16];
    __shared__ int   s_col[8][16];
    __shared__ float s_ext[8][16][5];
    __shared__ float s_diff[8][16][3];

    const int wave = threadIdx.x >> 5, lane = threadIdx.x & 31;
    const int M = lane & 15, half = lane >> 4, koff = half * 8;
    const int e0 = (blockIdx.x * 8 + wave) * 16;

    if (half == 0) {
        int e = e0 + lane;
        int r = eidx[e], c = eidx[E_TOT + e];
        s_row[wave][lane] = r;
        s_col[wave][lane] = c;
        float dx = pos[r * 3 + 0] - pos[c * 3 + 0];
        float dy = pos[r * 3 + 1] - pos[c * 3 + 1];
        float dz = pos[r * 3 + 2] - pos[c * 3 + 2];
        s_diff[wave][lane][0] = dx; s_diff[wave][lane][1] = dy; s_diff[wave][lane][2] = dz;
        s_ext[wave][lane][0] = dx * dx + dy * dy + dz * dz;
#pragma unroll
        for (int j = 0; j < 4; ++j) s_ext[wave][lane][1 + j] = edge_attr[(size_t)e * 4 + j];
    }
    __syncthreads();

    const unsigned short* rowR = hbf + (size_t)s_row[wave][M] * HID;
    const unsigned short* rowC = hbf + (size_t)s_col[wave][M] * HID;
    v16bf a[8];
#pragma unroll
    for (int kc = 0; kc < 4; ++kc) a[kc]     = load_afrag_bf(rowR, kc * 32 + koff);
#pragma unroll
    for (int kc = 0; kc < 4; ++kc) a[4 + kc] = load_afrag_bf(rowC, kc * 32 + koff);

    // ---- GEMM1: m1 = silu([hr|hc|radial|ea] @ e_w1 + e_b1)
#pragma unroll
    for (int nt = 0; nt < 8; ++nt) {
        int n = nt * 16 + M;
        float b = eb1[n];
        float wx[5];
#pragma unroll
        for (int j = 0; j < 5; ++j) wx[j] = ew1_full[(256 + j) * HID + n];
        v8f acc;
#pragma unroll
        for (int v = 0; v < 8; ++v) {
            int Mr = v + half * 8;
            float s = b;
#pragma unroll
            for (int j = 0; j < 5; ++j) s += s_ext[wave][Mr][j] * wx[j];
            acc[v] = s;
        }
#pragma unroll
        for (int kc = 0; kc < 8; ++kc) {
            if (kc == 0 && nt < 7)
                __builtin_prefetch(fr_ew1 + ((0) * 8 + nt + 1) * 512 + lane * 16, 0, 2);
            acc = wmma_bf16(a[kc], load_bfrag(fr_ew1 + (kc * 8 + nt) * 512, lane), acc);
        }
        acc = silu8(acc);
#pragma unroll
        for (int v = 0; v < 8; ++v)
            lds_m[wave][(v + half * 8) * HID + nt * 16 + M] = f2bf(acc[v]);
    }
    __syncthreads();

    // ---- GEMM2: m = silu(m1 @ e_w2 + e_b2); agg[row] += m; restage m
    v16bf a2[4];
#pragma unroll
    for (int kc = 0; kc < 4; ++kc)
        a2[kc] = load_afrag_bf(&lds_m[wave][M * HID], kc * 32 + koff);
    __syncthreads();
#pragma unroll
    for (int nt = 0; nt < 8; ++nt) {
        v8f acc; float b = eb2[nt * 16 + M];
#pragma unroll
        for (int v = 0; v < 8; ++v) acc[v] = b;
#pragma unroll
        for (int kc = 0; kc < 4; ++kc)
            acc = wmma_bf16(a2[kc], load_bfrag(fr_ew2 + (kc * 8 + nt) * 512, lane), acc);
        acc = silu8(acc);
#pragma unroll
        for (int v = 0; v < 8; ++v) {
            int Mr = v + half * 8;
            int node = s_row[wave][Mr];
            atomicAdd(&agg[(size_t)node * HID + nt * 16 + M], acc[v]);
            lds_m[wave][Mr * HID + nt * 16 + M] = f2bf(acc[v]);
        }
    }
    __syncthreads();

    // ---- GEMM3: t = silu(m @ c_w1 + c_b1); w = t @ c_w2 ; posnum[row] += diff*w
    v16bf a3[4];
#pragma unroll
    for (int kc = 0; kc < 4; ++kc)
        a3[kc] = load_afrag_bf(&lds_m[wave][M * HID], kc * 32 + koff);
    float part[8];
#pragma unroll
    for (int v = 0; v < 8; ++v) part[v] = 0.0f;
#pragma unroll
    for (int nt = 0; nt < 8; ++nt) {
        v8f acc; float b = cb1[nt * 16 + M];
#pragma unroll
        for (int v = 0; v < 8; ++v) acc[v] = b;
#pragma unroll
        for (int kc = 0; kc < 4; ++kc)
            acc = wmma_bf16(a3[kc], load_bfrag(fr_cw1 + (kc * 8 + nt) * 512, lane), acc);
        acc = silu8(acc);
        float wcol = cw2[nt * 16 + M];
#pragma unroll
        for (int v = 0; v < 8; ++v) part[v] += acc[v] * wcol;
    }
#pragma unroll
    for (int off = 1; off < 16; off <<= 1)
#pragma unroll
        for (int v = 0; v < 8; ++v) part[v] += __shfl_xor(part[v], off, 32);
    if (M < 8) {
        float wv = part[M];
        int Mr = M + half * 8;
        int node = s_row[wave][Mr];
#pragma unroll
        for (int d = 0; d < 3; ++d)
            atomicAdd(&posnum[(size_t)node * 3 + d], s_diff[wave][Mr][d] * wv);
    }
}

// ---------------------------------------------------------------- node update
__global__ __launch_bounds__(256) void node_kernel(
    float* __restrict__ h, unsigned short* __restrict__ hbf,
    const float* __restrict__ agg,
    const float* __restrict__ nb1, const float* __restrict__ nb2,
    const unsigned short* __restrict__ fr_nw1, const unsigned short* __restrict__ fr_nw2) {
    __shared__ alignas(16) unsigned short lds_m[8][16 * HID];
    const int wave = threadIdx.x >> 5, lane = threadIdx.x & 31;
    const int M = lane & 15, half = lane >> 4, koff = half * 8;
    const int n0 = (blockIdx.x * 8 + wave) * 16;

    const unsigned short* rowH = hbf + (size_t)(n0 + M) * HID;
    const float*          rowA = agg + (size_t)(n0 + M) * HID;
    v16bf a[8];
#pragma unroll
    for (int kc = 0; kc < 4; ++kc) a[kc]     = load_afrag_bf(rowH, kc * 32 + koff);
#pragma unroll
    for (int kc = 0; kc < 4; ++kc) a[4 + kc] = load_afrag_f32(rowA, kc * 32 + koff);

#pragma unroll
    for (int nt = 0; nt < 8; ++nt) {
        v8f acc; float b = nb1[nt * 16 + M];
#pragma unroll
        for (int v = 0; v < 8; ++v) acc[v] = b;
#pragma unroll
        for (int kc = 0; kc < 8; ++kc)
            acc = wmma_bf16(a[kc], load_bfrag(fr_nw1 + (kc * 8 + nt) * 512, lane), acc);
        acc = silu8(acc);
#pragma unroll
        for (int v = 0; v < 8; ++v)
            lds_m[wave][(v + half * 8) * HID + nt * 16 + M] = f2bf(acc[v]);
    }
    __syncthreads();
    v16bf a2[4];
#pragma unroll
    for (int kc = 0; kc < 4; ++kc)
        a2[kc] = load_afrag_bf(&lds_m[wave][M * HID], kc * 32 + koff);
#pragma unroll
    for (int nt = 0; nt < 8; ++nt) {
        v8f acc; float b = nb2[nt * 16 + M];
#pragma unroll
        for (int v = 0; v < 8; ++v) acc[v] = b;
#pragma unroll
        for (int kc = 0; kc < 4; ++kc)
            acc = wmma_bf16(a2[kc], load_bfrag(fr_nw2 + (kc * 8 + nt) * 512, lane), acc);
#pragma unroll
        for (int v = 0; v < 8; ++v) {
            size_t idx = (size_t)(n0 + v + half * 8) * HID + nt * 16 + M;
            float hv = h[idx] + acc[v];     // residual
            h[idx] = hv;
            hbf[idx] = f2bf(hv);
        }
    }
}

// ---------------------------------------------------------------- pos update
__global__ void pos_update_kernel(const float* __restrict__ posnum,
                                  const float* __restrict__ deg,
                                  float* __restrict__ pos_cur) {
    int i = blockIdx.x * blockDim.x + threadIdx.x;
    if (i >= N_NODES) return;
    float c = fmaxf(deg[i], 1.0f);
#pragma unroll
    for (int d = 0; d < 3; ++d)
        pos_cur[i * 3 + d] += posnum[i * 3 + d] / c;
}

// ---------------------------------------------------------------- generic K=128 WMMA GEMM (+bias)
__global__ __launch_bounds__(256) void gemm128_kernel(
    const unsigned short* __restrict__ inbf, int row0,
    const unsigned short* __restrict__ frW, const float* __restrict__ bias,
    float* __restrict__ out, unsigned short* __restrict__ outbf) {
    const int wave = threadIdx.x >> 5, lane = threadIdx.x & 31;
    const int M = lane & 15, half = lane >> 4, koff = half * 8;
    const int r0 = row0 + (blockIdx.x * 8 + wave) * 16;
    const unsigned short* rowp = inbf + (size_t)(r0 + M) * HID;
    v16bf a[4];
#pragma unroll
    for (int kc = 0; kc < 4; ++kc) a[kc] = load_afrag_bf(rowp, kc * 32 + koff);
#pragma unroll
    for (int nt = 0; nt < 8; ++nt) {
        v8f acc; float b = bias[nt * 16 + M];
#pragma unroll
        for (int v = 0; v < 8; ++v) acc[v] = b;
#pragma unroll
        for (int kc = 0; kc < 4; ++kc)
            acc = wmma_bf16(a[kc], load_bfrag(frW + (kc * 8 + nt) * 512, lane), acc);
#pragma unroll
        for (int v = 0; v < 8; ++v) {
            size_t idx = (size_t)(r0 + v + half * 8) * HID + nt * 16 + M;
            out[idx] = acc[v];
            if (outbf) outbf[idx] = f2bf(acc[v]);
        }
    }
}

// ---------------------------------------------------------------- GATv2 pieces
__global__ void self_logit_kernel(const float* gl_bl, const float* gl_br, const float* gl_att,
                                  const float* gp_bl, const float* gp_br, const float* gp_att,
                                  float* lself) {
    if (threadIdx.x == 0) {
        float a = 0.0f;
        for (int n = 0; n < HID; ++n) {
            float e = gl_bl[n] + gl_br[n];
            e = (e > 0.0f) ? e : 0.2f * e;
            a += e * gl_att[n];
        }
        lself[0] = a;
    } else if (threadIdx.x == 1) {
        float a = 0.0f;
        for (int n = 0; n < HID; ++n) {
            float e = gp_bl[n] + gp_br[n];
            e = (e > 0.0f) ? e : 0.2f * e;
            a += e * gp_att[n];
        }
        lself[1] = a;
    }
}

__global__ void gat_logit_kernel(const float* __restrict__ xl, const int* __restrict__ batch,
                                 int atom0, int natoms,
                                 const float* __restrict__ br, const float* __restrict__ att,
                                 float* __restrict__ logits, unsigned* __restrict__ maxb, int cbase) {
    int i = blockIdx.x * blockDim.x + threadIdx.x;
    if (i >= natoms) return;
    int atom = atom0 + i;
    float acc = 0.0f;
    for (int n = 0; n < HID; ++n) {
        float e = xl[(size_t)atom * HID + n] + br[n];
        e = (e > 0.0f) ? e : 0.2f * e;
        acc += e * att[n];
    }
    logits[atom] = acc;
    atomicMax(&maxb[cbase + batch[i]], fmap(acc));
}

__global__ __launch_bounds__(128) void gat_accum_kernel(
    const float* __restrict__ xl, const int* __restrict__ batch, int atom0,
    const float* __restrict__ logits, const unsigned* __restrict__ maxb, int cbase,
    const float* __restrict__ lselfbuf, int grp,
    float* __restrict__ den, float* __restrict__ num) {
    int i = blockIdx.x, n = threadIdx.x;
    int atom = atom0 + i;
    int c = cbase + batch[i];
    float m = fmaxf(funmap(maxb[c]), lselfbuf[grp]);
    float ex = __expf(logits[atom] - m);
    if (n == 0) atomicAdd(&den[c], ex);
    atomicAdd(&num[(size_t)c * HID + n], ex * xl[(size_t)atom * HID + n]);
}

__global__ void gat_final_kernel(const float* __restrict__ num, const float* __restrict__ den,
                                 const unsigned* __restrict__ maxb, int cbase, int C,
                                 const float* __restrict__ lselfbuf, int grp,
                                 const float* __restrict__ bl, const float* __restrict__ gbias,
                                 float* __restrict__ out) {
    int idx = blockIdx.x * blockDim.x + threadIdx.x;
    if (idx >= C * HID) return;
    int c = idx >> 7, n = idx & 127;
    float ls = lselfbuf[grp];
    float m = fmaxf(funmap(maxb[cbase + c]), ls);
    float exs = __expf(ls - m);
    float d = den[cbase + c] + exs;
    out[idx] = (num[(size_t)(cbase + c) * HID + n] + exs * bl[n]) / d + gbias[n];
}

__global__ void pos_out_kernel(const float* __restrict__ pos_cur, float* __restrict__ out) {
    int i = blockIdx.x * blockDim.x + threadIdx.x;
    if (i < SUM_LIG * 3) out[OUT_PLIG + i] = pos_cur[i];
    if (i < SUM_PRO * 3) out[OUT_PPRO + i] = pos_cur[SUM_LIG * 3 + i];
}

// ---------------------------------------------------------------- host
extern "C" void kernel_launch(void* const* d_in, const int* in_sizes, int n_in,
                              void* d_out, int out_size, void* d_ws, size_t ws_size,
                              hipStream_t stream) {
    (void)in_sizes; (void)n_in; (void)out_size; (void)ws_size;
    // inputs: setup_inputs order, params expanded in declaration order
    const float* x         = (const float*)d_in[0];
    const float* pos       = (const float*)d_in[1];
    const float* edge_attr = (const float*)d_in[2];
    const float* lin_w     = (const float*)d_in[3];
    const float* lin_b     = (const float*)d_in[4];
    const float* emb_in_w  = (const float*)d_in[5];
    const float* emb_in_b  = (const float*)d_in[6];
    const float* emb_out_w = (const float*)d_in[7];
    const float* emb_out_b = (const float*)d_in[8];
    const float* e_w1      = (const float*)d_in[9];
    const float* e_b1      = (const float*)d_in[10];
    const float* e_w2      = (const float*)d_in[11];
    const float* e_b2      = (const float*)d_in[12];
    const float* n_w1      = (const float*)d_in[13];
    const float* n_b1      = (const float*)d_in[14];
    const float* n_w2      = (const float*)d_in[15];
    const float* n_b2      = (const float*)d_in[16];
    const float* c_w1      = (const float*)d_in[17];
    const float* c_b1      = (const float*)d_in[18];
    const float* c_w2      = (const float*)d_in[19];
    const float* gl_wl     = (const float*)d_in[20];
    const float* gl_bl     = (const float*)d_in[21];
    const float* gl_br     = (const float*)d_in[23];
    const float* gl_att    = (const float*)d_in[24];
    const float* gl_bias   = (const float*)d_in[25];
    const float* gp_wl     = (const float*)d_in[26];
    const float* gp_bl     = (const float*)d_in[27];
    const float* gp_br     = (const float*)d_in[29];
    const float* gp_att    = (const float*)d_in[30];
    const float* gp_bias   = (const float*)d_in[31];
    const int* edge_index  = (const int*)d_in[32];
    const int* batch_lig   = (const int*)d_in[33];
    const int* batch_pro   = (const int*)d_in[34];
    float* out = (float*)d_out;

    // workspace carve-up
    char* ws = (char*)d_ws;
    size_t off = 0;
    auto alloc = [&](size_t bytes) -> void* {
        void* p = ws + off;
        off = (off + bytes + 255) & ~(size_t)255;
        return p;
    };
    float*          h       = (float*)alloc((size_t)N_NODES * HID * 4);
    unsigned short* hbf     = (unsigned short*)alloc((size_t)N_NODES * HID * 2);
    float*          agg     = (float*)alloc((size_t)N_NODES * HID * 4);
    float*          posnum  = (float*)alloc((size_t)N_NODES * 3 * 4);
    float*          pos_cur = (float*)alloc((size_t)N_NODES * 3 * 4);
    float*          deg     = (float*)alloc((size_t)N_NODES * 4);
    float*          hfin    = (float*)alloc((size_t)N_NODES * HID * 4);
    unsigned short* hfinbf  = (unsigned short*)alloc((size_t)N_NODES * HID * 2);
    float*          xl      = (float*)alloc((size_t)N_NODES * HID * 4);
    float*          logits  = (float*)alloc((size_t)N_NODES * 4);
    unsigned*       maxb    = (unsigned*)alloc((size_t)(C_LIG + C_PRO) * 4);
    float*          den     = (float*)alloc((size_t)(C_LIG + C_PRO) * 4);
    float*          num     = (float*)alloc((size_t)(C_LIG + C_PRO) * HID * 4);
    float*          lself   = (float*)alloc(2 * 4);
    unsigned short* fr_lin    = (unsigned short*)alloc((size_t)2 * 8 * 512 * 2);
    unsigned short* fr_embin  = (unsigned short*)alloc((size_t)4 * 8 * 512 * 2);
    unsigned short* fr_embout = (unsigned short*)alloc((size_t)4 * 8 * 512 * 2);
    unsigned short* fr_glwl   = (unsigned short*)alloc((size_t)4 * 8 * 512 * 2);
    unsigned short* fr_gpwl   = (unsigned short*)alloc((size_t)4 * 8 * 512 * 2);
    unsigned short* fr_ew1    = (unsigned short*)alloc((size_t)N_LAYERS * 8 * 8 * 512 * 2);
    unsigned short* fr_ew2    = (unsigned short*)alloc((size_t)N_LAYERS * 4 * 8 * 512 * 2);
    unsigned short* fr_cw1    = (unsigned short*)alloc((size_t)N_LAYERS * 4 * 8 * 512 * 2);
    unsigned short* fr_nw1    = (unsigned short*)alloc((size_t)N_LAYERS * 8 * 8 * 512 * 2);
    unsigned short* fr_nw2    = (unsigned short*)alloc((size_t)N_LAYERS * 4 * 8 * 512 * 2);

    auto prep = [&](const float* src, int Ksrc, unsigned short* dst, int KC) {
        prep_frag_kernel<<<dim3(KC * 8), dim3(256), 0, stream>>>(src, Ksrc, dst);
    };

    // ---- weight fragment prep (bf16, B-fragment layout; stays hot in 192MB L2)
    prep(lin_w, 44, fr_lin, 2);
    prep(emb_in_w, 128, fr_embin, 4);
    prep(emb_out_w, 128, fr_embout, 4);
    prep(gl_wl, 128, fr_glwl, 4);
    prep(gp_wl, 128, fr_gpwl, 4);
    for (int l = 0; l < N_LAYERS; ++l) {
        prep(e_w1 + (size_t)l * 261 * HID, 256, fr_ew1 + (size_t)l * 8 * 8 * 512, 8);
        prep(e_w2 + (size_t)l * HID * HID, 128, fr_ew2 + (size_t)l * 4 * 8 * 512, 4);
        prep(c_w1 + (size_t)l * HID * HID, 128, fr_cw1 + (size_t)l * 4 * 8 * 512, 4);
        prep(n_w1 + (size_t)l * 256 * HID, 256, fr_nw1 + (size_t)l * 8 * 8 * 512, 8);
        prep(n_w2 + (size_t)l * HID * HID, 128, fr_nw2 + (size_t)l * 4 * 8 * 512, 4);
    }

    hipMemcpyAsync(pos_cur, pos, (size_t)N_NODES * 3 * 4, hipMemcpyDeviceToDevice, stream);
    hipMemsetAsync(deg, 0, (size_t)N_NODES * 4, stream);
    degree_kernel<<<(E_TOT + 255) / 256, 256, 0, stream>>>(edge_index, deg);
    embed_kernel<<<N_NODES / 128, 256, 0, stream>>>(x, fr_lin, lin_b, fr_embin, emb_in_b, h, hbf);

    // ---- EGNN layers
    for (int l = 0; l < N_LAYERS; ++l) {
        hipMemsetAsync(agg, 0, (size_t)N_NODES * HID * 4, stream);
        hipMemsetAsync(posnum, 0, (size_t)N_NODES * 3 * 4, stream);
        edge_kernel<<<E_TOT / 128, 256, 0, stream>>>(
            hbf, pos_cur, edge_index, edge_attr,
            e_w1 + (size_t)l * 261 * HID,
            e_b1 + l * HID, e_b2 + l * HID, c_b1 + l * HID, c_w2 + l * HID,
            fr_ew1 + (size_t)l * 8 * 8 * 512,
            fr_ew2 + (size_t)l * 4 * 8 * 512,
            fr_cw1 + (size_t)l * 4 * 8 * 512,
            agg, posnum);
        node_kernel<<<N_NODES / 128, 256, 0, stream>>>(
            h, hbf, agg, n_b1 + l * HID, n_b2 + l * HID,
            fr_nw1 + (size_t)l * 8 * 8 * 512, fr_nw2 + (size_t)l * 4 * 8 * 512);
        pos_update_kernel<<<(N_NODES + 255) / 256, 256, 0, stream>>>(posnum, deg, pos_cur);
    }

    // ---- emb_out
    gemm128_kernel<<<N_NODES / 128, 256, 0, stream>>>(hbf, 0, fr_embout, emb_out_b, hfin, hfinbf);

    // ---- GATv2 pooling (lig / pro)
    gemm128_kernel<<<SUM_LIG / 128, 256, 0, stream>>>(hfinbf, 0, fr_glwl, gl_bl, xl, (unsigned short*)nullptr);
    gemm128_kernel<<<SUM_PRO / 128, 256, 0, stream>>>(hfinbf, SUM_LIG, fr_gpwl, gp_bl, xl, (unsigned short*)nullptr);
    self_logit_kernel<<<1, 2, 0, stream>>>(gl_bl, gl_br, gl_att, gp_bl, gp_br, gp_att, lself);
    hipMemsetAsync(maxb, 0, (size_t)(C_LIG + C_PRO) * 4, stream);
    gat_logit_kernel<<<SUM_LIG / 256, 256, 0, stream>>>(xl, batch_lig, 0, SUM_LIG, gl_br, gl_att, logits, maxb, 0);
    gat_logit_kernel<<<SUM_PRO / 256, 256, 0, stream>>>(xl, batch_pro, SUM_LIG, SUM_PRO, gp_br, gp_att, logits, maxb, C_LIG);
    hipMemsetAsync(den, 0, (size_t)(C_LIG + C_PRO) * 4, stream);
    hipMemsetAsync(num, 0, (size_t)(C_LIG + C_PRO) * HID * 4, stream);
    gat_accum_kernel<<<SUM_LIG, 128, 0, stream>>>(xl, batch_lig, 0, logits, maxb, 0, lself, 0, den, num);
    gat_accum_kernel<<<SUM_PRO, 128, 0, stream>>>(xl, batch_pro, SUM_LIG, logits, maxb, C_LIG, lself, 1, den, num);
    gat_final_kernel<<<(C_LIG * HID + 255) / 256, 256, 0, stream>>>(num, den, maxb, 0, C_LIG, lself, 0, gl_bl, gl_bias, out);
    gat_final_kernel<<<(C_PRO * HID + 255) / 256, 256, 0, stream>>>(num, den, maxb, C_LIG, C_PRO, lself, 1, gp_bl, gp_bias, out + OUT_XPRO);

    // ---- pos outputs
    pos_out_kernel<<<(SUM_PRO * 3 + 255) / 256, 256, 0, stream>>>(pos_cur, out);
}
